// ProductQuantizer_26087631356135
// MI455X (gfx1250) — matrix operand order repaired and checked by reading, not compiled, not run on hardware
//
#include <hip/hip_runtime.h>
#include <hip/hip_bf16.h>

// ---------------- problem constants (from reference) ----------------
#define BDIM 16
#define NDIM 4096
#define FDIM 512
#define SDIM 8
#define KDIM 1024
#define DDIM 64
#define TOK  (BDIM * NDIM)            // 65536 tokens
#define QTOT ((size_t)TOK * FDIM)     // 33554432 floats of quant output

// ws layout (floats): [0..15] pad, [16 .. 16+8191] h2 = 0.5*||c||^2,
//                     [8208 .. 8208+4095] per-block dev partials
#define H2_OFF   16
#define PART_OFF (16 + SDIM * KDIM)
#define NPART    (SDIM * (TOK / 128))  // 8 * 512 = 4096 blocks

typedef __attribute__((ext_vector_type(16))) __bf16 v16bf;
typedef __attribute__((ext_vector_type(8)))  float  v8f;
typedef __attribute__((ext_vector_type(2)))  float  f2v;   // native vec2 (NT-store capable)

// LDS codebook chunk: 256 codes x 64 bf16, row stride padded to 36 dwords
// (gcd(36,64)=4 -> 16 lanes * stride 36 cover 16 distinct bank groups)
#define CHUNK        256
#define NCHUNK       (KDIM / CHUNK)    // 4
#define TILES_PER_CH (CHUNK / 16)      // 16
#define ROWU         36                // dwords per code row in LDS (32 used + 4 pad)

__device__ __forceinline__ unsigned short f2bf_bits(float f) {
    unsigned u = __builtin_bit_cast(unsigned, f);
    unsigned r = u + 0x7FFFu + ((u >> 16) & 1u);   // round-to-nearest-even
    return (unsigned short)(r >> 16);
}
__device__ __forceinline__ __bf16 f2bf(float f) {
    unsigned short b = f2bf_bits(f);
    return __builtin_bit_cast(__bf16, b);
}
__device__ __forceinline__ v16bf load_b16(const unsigned int* p) {
    union { uint4 q[2]; v16bf v; } t;
    const uint4* p4 = (const uint4*)p;   // p is 16B aligned by construction
    t.q[0] = p4[0];
    t.q[1] = p4[1];
    return t.v;
}

// ---------------- kernel 0: h2 table = 0.5 * ||codebook_k||^2 ----------------
__global__ void pq_prep(const float* __restrict__ cb, float* __restrict__ ws) {
    int k = blockIdx.x * blockDim.x + threadIdx.x;   // 0 .. S*K-1
    if (k < SDIM * KDIM) {
        const float* row = cb + (size_t)k * DDIM;
        float s = 0.f;
#pragma unroll 8
        for (int d = 0; d < DDIM; ++d) { float v = row[d]; s += v * v; }
        ws[H2_OFF + k] = 0.5f * s;
    }
}

// ---------------- kernel 1: fused WMMA scores + argmin + gather ----------------
__global__ __launch_bounds__(256)
void pq_main(const float* __restrict__ x, const float* __restrict__ cb,
             float* __restrict__ outq, float* __restrict__ ws) {
    __shared__ __align__(16) unsigned int lds_cb[CHUNK * ROWU];  // ~36.9 KB
    __shared__ float lds_h2[CHUNK];
    __shared__ int   lds_idx[8 * 16];
    __shared__ float lds_dev[8];

    const int s    = blockIdx.y;
    const int w    = threadIdx.x >> 5;
    const int lane = threadIdx.x & 31;
    const int half = lane >> 4;          // 0: lanes 0-15, 1: lanes 16-31
    const int l    = lane & 15;
    const int tokBase = blockIdx.x * 128 + w * 16;

    // ---- A: 16 tokens x 64 dims, bf16, contiguous k order (same as B) ----
    // lanes 0-15: elems 0-15 = K{0..15}
    // lanes16-31: elems 0-15 = K{16..31}
    const float* xrow = x + (size_t)(tokBase + l) * FDIM + s * DDIM;
    v16bf a0, a1;
    {
        const int lo = 16 * half, hi = 16 * half + 8;
#pragma unroll
        for (int j = 0; j < 8; ++j) {
            a0[j]     = f2bf(xrow[lo + j]);
            a0[8 + j] = f2bf(xrow[hi + j]);
            a1[j]     = f2bf(xrow[32 + lo + j]);
            a1[8 + j] = f2bf(xrow[32 + hi + j]);
        }
    }

    float best[8];
    int   bidx[8];
#pragma unroll
    for (int r = 0; r < 8; ++r) { best[r] = -3.4e38f; bidx[r] = 0; }

    for (int c = 0; c < NCHUNK; ++c) {
        __syncthreads();   // previous chunk fully consumed
        // stage 256 codes x 64 f32 -> bf16 into LDS (padded rows) + h2 slice
        const float4* src = (const float4*)(cb + ((size_t)s * KDIM + c * CHUNK) * DDIM);
#pragma unroll
        for (int j = 0; j < 16; ++j) {
            int f = threadIdx.x + 256 * j;        // float4 index, 4096 per chunk
            float4 v = src[f];
            int row = f >> 4;                     // 16 float4 per code row
            int col = (f & 15) * 2;               // dword pair within row
            lds_cb[row * ROWU + col]     = (unsigned)f2bf_bits(v.x) | ((unsigned)f2bf_bits(v.y) << 16);
            lds_cb[row * ROWU + col + 1] = (unsigned)f2bf_bits(v.z) | ((unsigned)f2bf_bits(v.w) << 16);
        }
        lds_h2[threadIdx.x] = ws[H2_OFF + s * KDIM + c * CHUNK + threadIdx.x];
        __syncthreads();

        // prefetch next chunk into cache while we compute on this one
        if (c + 1 < NCHUNK) {
            const char* nsrc = (const char*)(cb + ((size_t)s * KDIM + (c + 1) * CHUNK) * DDIM);
            __builtin_prefetch(nsrc + (size_t)threadIdx.x * 256, 0, 0);  // 256 thr * 256B = 64KB
        }

        // two code tiles per iteration: independent WMMA chains fill each
        // other's WMMA->VALU hazard slots
        for (int t = 0; t < TILES_PER_CH; t += 2) {
            const int code0 = c * CHUNK + t * 16 + l;
            const int code1 = code0 + 16;
            const unsigned int* brow0 = lds_cb + (t * 16 + l) * ROWU;
            const unsigned int* brow1 = brow0 + 16 * ROWU;
            v16bf p0 = load_b16(brow0 + 8 * half);        // tile t,   d = 16*half ..
            v16bf p1 = load_b16(brow0 + 16 + 8 * half);   // tile t,   d = 32+16*half ..
            v16bf q0 = load_b16(brow1 + 8 * half);        // tile t+1
            v16bf q1 = load_b16(brow1 + 16 + 8 * half);
            v8f acc0 = {};
            v8f acc1 = {};
            acc0 = __builtin_amdgcn_wmma_f32_16x16x32_bf16(false, a0, false, p0,
                                                           (short)0, acc0, false, false);
            acc1 = __builtin_amdgcn_wmma_f32_16x16x32_bf16(false, a0, false, q0,
                                                           (short)0, acc1, false, false);
            acc0 = __builtin_amdgcn_wmma_f32_16x16x32_bf16(false, a1, false, p1,
                                                           (short)0, acc0, false, false);
            acc1 = __builtin_amdgcn_wmma_f32_16x16x32_bf16(false, a1, false, q1,
                                                           (short)0, acc1, false, false);
            const float h0 = lds_h2[t * 16 + l];
            const float h1 = lds_h2[t * 16 + 16 + l];
#pragma unroll
            for (int r = 0; r < 8; ++r) {
                float sc0 = acc0[r] - h0;                 // argmax(xc - 0.5*c2) == argmin dist
                if (sc0 > best[r]) { best[r] = sc0; bidx[r] = code0; }
                float sc1 = acc1[r] - h1;
                if (sc1 > best[r]) { best[r] = sc1; bidx[r] = code1; }
            }
        }
    }

    // ---- cross-lane argmax within each 16-lane column group ----
#pragma unroll
    for (int r = 0; r < 8; ++r) {
        float bv = best[r]; int bi = bidx[r];
#pragma unroll
        for (int m = 1; m < 16; m <<= 1) {
            float ov = __shfl_xor(bv, m, 32);
            int   oi = __shfl_xor(bi, m, 32);
            if (ov > bv || (ov == bv && oi < bi)) { bv = ov; bi = oi; }  // first-min tie-break
        }
        best[r] = bv; bidx[r] = bi;
    }
    if (l == 0) {
#pragma unroll
        for (int r = 0; r < 8; ++r) lds_idx[w * 16 + half * 8 + r] = bidx[r];  // token M = r + 8*half
    }
    __syncthreads();

    // ---- gather codewords, write quant (NT stores), accumulate (sym - xs)^2 ----
    float devAcc = 0.f;
    for (int t = 0; t < 16; ++t) {
        const int row  = tokBase + t;
        const int kidx = lds_idx[w * 16 + t];
        const f2v* crow = (const f2v*)(cb + ((size_t)s * KDIM + kidx) * DDIM);
        f2v v = crow[lane];                                        // d = 2*lane, 2*lane+1
        const size_t off = ((size_t)row * FDIM + s * DDIM) / 2 + lane;
        f2v xv = ((const f2v*)x)[off];
        __builtin_nontemporal_store(v, (f2v*)outq + off);          // quant is write-once
        float d0 = v.x - xv.x, d1 = v.y - xv.y;
        devAcc += d0 * d0 + d1 * d1;
    }
    for (int m = 16; m; m >>= 1) devAcc += __shfl_xor(devAcc, m, 32);
    if (lane == 0) lds_dev[w] = devAcc;
    __syncthreads();
    if (threadIdx.x == 0) {
        float blk = 0.f;
#pragma unroll
        for (int i = 0; i < 8; ++i) blk += lds_dev[i];
        ws[PART_OFF + blockIdx.y * gridDim.x + blockIdx.x] = blk;  // deterministic partial
    }
}

// ---------------- kernel 2: deterministic final reduction ----------------
__global__ void pq_final(const float* __restrict__ ws, float* __restrict__ outq) {
    if (threadIdx.x == 0 && blockIdx.x == 0) {
        double acc = 0.0;
        for (int i = 0; i < NPART; ++i) acc += (double)ws[PART_OFF + i];
        // dev = (1 + 0.25) * total_sq / (B*N*D); B*N*D = 4194304
        outq[QTOT] = (float)(1.25 * acc / 4194304.0);
    }
}

extern "C" void kernel_launch(void* const* d_in, const int* in_sizes, int n_in,
                              void* d_out, int out_size, void* d_ws, size_t ws_size,
                              hipStream_t stream) {
    (void)in_sizes; (void)n_in; (void)out_size; (void)ws_size;
    const float* x  = (const float*)d_in[0];   // (B, N, F) f32
    const float* cb = (const float*)d_in[1];   // (S, K, D) f32
    float* outq = (float*)d_out;               // quant (B*N*F) then dev scalar
    float* ws   = (float*)d_ws;

    pq_prep <<<dim3((SDIM * KDIM + 255) / 256), dim3(256), 0, stream>>>(cb, ws);
    pq_main <<<dim3(TOK / 128, SDIM),           dim3(256), 0, stream>>>(x, cb, outq, ws);
    pq_final<<<dim3(1),                         dim3(1),   0, stream>>>(ws, outq);
}
